// WindowAttention_63608465654183
// MI455X (gfx1250) — matrix operand order repaired and checked by reading, not compiled, not run on hardware
//
#include <hip/hip_runtime.h>
#include <hip/hip_bf16.h>

// ---------------------------------------------------------------------------
// Shifted-window attention (Swin), fused per-window kernel for gfx1250.
// bf16 WMMA 16x16x32 with f32 accumulate for all four matmul stages.
// ---------------------------------------------------------------------------

typedef __bf16 bf16;
typedef __attribute__((ext_vector_type(16))) __bf16 bf16x16;
typedef __attribute__((ext_vector_type(8)))  __bf16 bf16x8;
typedef __attribute__((ext_vector_type(8)))  float  f32x8;

#define WMMA_BF16(A, B, C) \
  __builtin_amdgcn_wmma_f32_16x16x32_bf16(false, (A), false, (B), (short)0, (C), false, false)

constexpr int WSZ   = 8;
constexpr int L     = 64;     // WSZ*WSZ tokens per window
constexpr int DIM   = 256;
constexpr int NH    = 8;
constexpr int HD    = 32;
constexpr int IMG   = 128;    // H == W
constexpr int SHIFT = 4;
constexpr float SCALE = 0.17677669529663687f;  // HD^-0.5

__device__ __forceinline__ f32x8 zero8() {
  f32x8 z;
#pragma unroll
  for (int i = 0; i < 8; ++i) z[i] = 0.0f;
  return z;
}

__device__ __forceinline__ bf16x16 frag_combine(bf16x8 lo, bf16x8 hi) {
  bf16x16 r;
#pragma unroll
  for (int i = 0; i < 8; ++i) { r[i] = lo[i]; r[i + 8] = hi[i]; }
  return r;
}

// A fragment (16x32 bf16) from row-major [m][k] source.
// ISA layout: lane<16 holds K in {8*0..7, 16+0..7}? per-half: lane half hi
// owns K = [8*hi, 8*hi+8) in v0..3 and [16+8*hi, 16+8*hi+8) in v4..7.
__device__ __forceinline__ bf16x16 load_a(const bf16* base, int stride, int lane) {
  const int m = lane & 15, hi = lane >> 4;
  const bf16* p = base + m * stride + hi * 8;
  bf16x8 lo = *(const bf16x8*)(p);
  bf16x8 h2 = *(const bf16x8*)(p + 16);
  return frag_combine(lo, h2);
}

// B fragment (32x16 bf16) from transposed [n][k] source (stride = K extent).
// lane half hi owns K = [8*hi, 8*hi+8) in v0..3 and [16+8*hi, 16+8*hi+8) in v4..7 for column n = lane&15.
__device__ __forceinline__ bf16x16 load_b(const bf16* baseT, int stride, int lane) {
  const int n = lane & 15, hi = lane >> 4;
  const bf16* p = baseT + n * stride + hi * 8;
  bf16x8 lo = *(const bf16x8*)(p);
  bf16x8 h2 = *(const bf16x8*)(p + 16);
  return frag_combine(lo, h2);
}

// shift-mask region category along one axis (shifted coordinates)
__device__ __forceinline__ int catc(int g) { return (g < IMG - WSZ) ? 0 : ((g < IMG - SHIFT) ? 1 : 2); }

// ---------------------------------------------------------------------------
// Weight prep: fp32 row-major [K][N] -> bf16 transposed [N][K]
// ---------------------------------------------------------------------------
__global__ void __launch_bounds__(256) cvt_transpose_kernel(const float* __restrict__ src,
                                                            bf16* __restrict__ dst,
                                                            int K, int N) {
  int e = blockIdx.x * 256 + threadIdx.x;
  if (e < K * N) {
    int k = e / N, n = e - k * N;
    dst[n * K + k] = (bf16)src[e];
  }
}

// ---------------------------------------------------------------------------
// Fused window-attention kernel: 1 block = 1 window, 256 threads = 8 waves
// ---------------------------------------------------------------------------
__global__ void __launch_bounds__(256) swin_attn_kernel(
    const float* __restrict__ x, const float* __restrict__ y,
    const float* __restrict__ kv_b, const float* __restrict__ q_b,
    const float* __restrict__ proj_b, const float* __restrict__ bias_table,
    const bf16* __restrict__ kvT, const bf16* __restrict__ qT,
    const bf16* __restrict__ projT, float* __restrict__ out) {
  extern __shared__ __align__(16) char smem[];
  bf16* s_x  = (bf16*)smem;          // [L][DIM]  32KB (kv source; later P scratch lo)
  bf16* s_y  = s_x + L * DIM;        // [L][DIM]  32KB (q source;  later P scratch hi)
  bf16* s_q  = s_y + L * DIM;        // [NH][L][HD] 32KB (later s_o = attn out [L][DIM])
  bf16* s_k  = s_q + L * DIM;        // [NH][L][HD] 32KB
  bf16* s_vt = s_k + L * DIM;        // [NH][HD][L] 32KB (v transposed)
  float* s_bias = (float*)(s_vt + L * DIM);  // [225][NH] 7.2KB
  bf16* p_scr = s_x;                 // 64KB overlay for per-wave P (8KB each)
  bf16* s_o   = s_q;                 // overlay: attention output [L][DIM]

  const int tid  = threadIdx.x;
  const int wave = tid >> 5, lane = tid & 31;
  const int wid  = blockIdx.x;
  const int b0 = wid >> 8, wloc = wid & 255, wr = wloc >> 4, wc = wloc & 15;

  // ---- phase 0: stage shifted window (bf16) + bias table -------------------
  for (int i = tid; i < 225 * NH; i += 256) s_bias[i] = bias_table[i];
  for (int i = tid; i < L * (DIM / 4); i += 256) {
    int t = i >> 6, cv = (i & 63) * 4;
    int lr = t >> 3, lc = t & 7;
    int hh = (wr * WSZ + lr + SHIFT) & (IMG - 1);
    int ww = (wc * WSZ + lc + SHIFT) & (IMG - 1);
    size_t g = ((size_t)(b0 * IMG + hh) * IMG + ww) * DIM + cv;
    float4 vx = *(const float4*)(x + g);
    float4 vy = *(const float4*)(y + g);
    bf16* px = s_x + t * DIM + cv;
    bf16* py = s_y + t * DIM + cv;
    px[0] = (bf16)vx.x; px[1] = (bf16)vx.y; px[2] = (bf16)vx.z; px[3] = (bf16)vx.w;
    py[0] = (bf16)vy.x; py[1] = (bf16)vy.y; py[2] = (bf16)vy.z; py[3] = (bf16)vy.w;
  }
  __syncthreads();

  // ---- phase 1: q = (yw@q_w + b)*scale ; k,v = xw@kv_w + b -----------------
  // 64 q tiles + 128 kv tiles of 16x16, K=256 in 8 WMMA steps, 8 waves.
  for (int tile = wave; tile < 192; tile += 8) {
    const bool isq = tile < 64;
    int id = isq ? tile : (tile - 64);
    int mt = id & 3, nt = id >> 2;
    const bf16* src = isq ? s_y : s_x;
    const bf16* wT  = isq ? qT : kvT;
    f32x8 acc = zero8();
#pragma unroll
    for (int kk = 0; kk < 8; ++kk) {
      bf16x16 a = load_a(src + mt * 16 * DIM + kk * 32, DIM, lane);
      bf16x16 b = load_b(wT + (nt * 16) * DIM + kk * 32, DIM, lane);
      acc = WMMA_BF16(a, b, acc);
    }
    const int n = nt * 16 + (lane & 15);
    const int mhi = (lane >> 4) * 8;  // D layout: M = r + 8*(lane>=16)
    if (isq) {
      float bb = q_b[n];
      int head = n >> 5, hd = n & 31;
#pragma unroll
      for (int r = 0; r < 8; ++r) {
        int m = mt * 16 + r + mhi;
        s_q[head * (L * HD) + m * HD + hd] = (bf16)((acc[r] + bb) * SCALE);
      }
    } else {
      float bb = kv_b[n];
      if (n < DIM) {  // k part
        int head = n >> 5, hd = n & 31;
#pragma unroll
        for (int r = 0; r < 8; ++r) {
          int m = mt * 16 + r + mhi;
          s_k[head * (L * HD) + m * HD + hd] = (bf16)(acc[r] + bb);
        }
      } else {        // v part, stored transposed [hd][token]
        int fv = n - DIM;
        int head = fv >> 5, hd = fv & 31;
#pragma unroll
        for (int r = 0; r < 8; ++r) {
          int m = mt * 16 + r + mhi;
          s_vt[head * (HD * L) + hd * L + m] = (bf16)(acc[r] + bb);
        }
      }
    }
  }
  __syncthreads();

  // ---- phase 2: per-head S = q k^T, +bias +mask, softmax+1, P -> LDS -------
  {
    const int h = wave;
    const bf16* qh = s_q + h * L * HD;
    const bf16* kh = s_k + h * L * HD;
    bf16* pw = p_scr + h * L * L;
    const int N = lane & 15, hi = lane >> 4;
    for (int mt = 0; mt < 4; ++mt) {
      bf16x16 a = load_a(qh + mt * 16 * HD, HD, lane);  // K = 32 = HD, one step
      f32x8 s[4];
#pragma unroll
      for (int nt = 0; nt < 4; ++nt) {
        bf16x16 b = load_b(kh + (nt * 16) * HD, HD, lane);
        f32x8 z = zero8();
        s[nt] = WMMA_BF16(a, b, z);
      }
      // relative-position bias + shift mask
#pragma unroll
      for (int nt = 0; nt < 4; ++nt) {
        int j = nt * 16 + N;
        int rj = j >> 3, cj = j & 7;
        int labj = catc(wr * WSZ + rj) * 3 + catc(wc * WSZ + cj);
#pragma unroll
        for (int r = 0; r < 8; ++r) {
          int i = mt * 16 + r + hi * 8;
          int ri = i >> 3, ci = i & 7;
          int idx = (ri - rj + 7) * 15 + (ci - cj + 7);
          float v = s[nt][r] + s_bias[idx * NH + h];
          int labi = catc(wr * WSZ + ri) * 3 + catc(wc * WSZ + ci);
          if (labi != labj) v -= 100.0f;
          s[nt][r] = v;
        }
      }
      // softmax-plus-one per row (row lives in one 16-lane half of the wave)
#pragma unroll
      for (int r = 0; r < 8; ++r) {
        float mx = fmaxf(fmaxf(s[0][r], s[1][r]), fmaxf(s[2][r], s[3][r]));
#pragma unroll
        for (int d = 8; d >= 1; d >>= 1) mx = fmaxf(mx, __shfl_xor(mx, d, 16));
        float sum = 0.0f;
#pragma unroll
        for (int nt = 0; nt < 4; ++nt) { float e = __expf(s[nt][r] - mx); s[nt][r] = e; sum += e; }
#pragma unroll
        for (int d = 8; d >= 1; d >>= 1) sum += __shfl_xor(sum, d, 16);
        float inv = 1.0f / (sum + 1.0f);
#pragma unroll
        for (int nt = 0; nt < 4; ++nt) s[nt][r] *= inv;
      }
      // store P (bf16) row-major [i][j] for A-fragment reload
#pragma unroll
      for (int nt = 0; nt < 4; ++nt) {
        int j = nt * 16 + N;
#pragma unroll
        for (int r = 0; r < 8; ++r) {
          int i = mt * 16 + r + hi * 8;
          pw[i * L + j] = (bf16)s[nt][r];
        }
      }
    }
  }
  __syncthreads();

  // ---- phase 3: out_h = P @ v (per head), pack into s_o[L][DIM] ------------
  {
    const int h = wave;
    const bf16* vt = s_vt + h * HD * L;
    const bf16* pw = p_scr + h * L * L;
    for (int mt = 0; mt < 4; ++mt) {
      for (int nt = 0; nt < 2; ++nt) {
        f32x8 acc = zero8();
#pragma unroll
        for (int kk = 0; kk < 2; ++kk) {
          bf16x16 a = load_a(pw + mt * 16 * L + kk * 32, L, lane);
          bf16x16 b = load_b(vt + (nt * 16) * L + kk * 32, L, lane);
          acc = WMMA_BF16(a, b, acc);
        }
        const int N = lane & 15, mhi = (lane >> 4) * 8;
        const int c = h * HD + nt * 16 + N;
#pragma unroll
        for (int r = 0; r < 8; ++r)
          s_o[(mt * 16 + r + mhi) * DIM + c] = (bf16)acc[r];
      }
    }
  }
  __syncthreads();

  // ---- phase 4: out = s_o @ proj_w + b ; window-reverse + reverse-roll -----
  for (int tile = wave; tile < 64; tile += 8) {
    int mt = tile >> 4, nt = tile & 15;
    f32x8 acc = zero8();
#pragma unroll
    for (int kk = 0; kk < 8; ++kk) {
      bf16x16 a = load_a(s_o + mt * 16 * DIM + kk * 32, DIM, lane);
      bf16x16 b = load_b(projT + (nt * 16) * DIM + kk * 32, DIM, lane);
      acc = WMMA_BF16(a, b, acc);
    }
    const int N = lane & 15, mhi = (lane >> 4) * 8;
    const int c = nt * 16 + N;
    float pb = proj_b[c];
#pragma unroll
    for (int r = 0; r < 8; ++r) {
      int t = mt * 16 + r + mhi;
      int lr = t >> 3, lc = t & 7;
      int hh = (wr * WSZ + lr + SHIFT) & (IMG - 1);
      int ww = (wc * WSZ + lc + SHIFT) & (IMG - 1);
      out[((size_t)(b0 * IMG + hh) * IMG + ww) * DIM + c] = acc[r] + pb;
    }
  }
}

// ---------------------------------------------------------------------------
extern "C" void kernel_launch(void* const* d_in, const int* in_sizes, int n_in,
                              void* d_out, int out_size, void* d_ws, size_t ws_size,
                              hipStream_t stream) {
  const float* x          = (const float*)d_in[0];
  const float* y          = (const float*)d_in[1];
  const float* kv_w       = (const float*)d_in[2];
  const float* kv_b       = (const float*)d_in[3];
  const float* q_w        = (const float*)d_in[4];
  const float* q_b        = (const float*)d_in[5];
  const float* proj_w     = (const float*)d_in[6];
  const float* proj_b     = (const float*)d_in[7];
  const float* bias_table = (const float*)d_in[8];
  float* out = (float*)d_out;

  // workspace: bf16 transposed weights
  bf16* kvT   = (bf16*)d_ws;                 // [2*DIM][DIM] = 512*256
  bf16* qT    = kvT + 2 * DIM * DIM;         // [DIM][DIM]
  bf16* projT = qT + DIM * DIM;              // [DIM][DIM]

  cvt_transpose_kernel<<<(DIM * 2 * DIM + 255) / 256, 256, 0, stream>>>(kv_w, kvT, DIM, 2 * DIM);
  cvt_transpose_kernel<<<(DIM * DIM + 255) / 256, 256, 0, stream>>>(q_w, qT, DIM, DIM);
  cvt_transpose_kernel<<<(DIM * DIM + 255) / 256, 256, 0, stream>>>(proj_w, projT, DIM, DIM);

  constexpr size_t SMEM = (size_t)(5 * L * DIM) * sizeof(bf16) + 225 * NH * sizeof(float);
  (void)hipFuncSetAttribute((const void*)swin_attn_kernel,
                            hipFuncAttributeMaxDynamicSharedMemorySize, (int)SMEM);

  const int nWindows = 16 * (IMG / WSZ) * (IMG / WSZ);  // 4096
  swin_attn_kernel<<<nWindows, 256, SMEM, stream>>>(
      x, y, kv_b, q_b, proj_b, bias_table, kvT, qT, projT, out);
}
